// MultiScaleConvOp_45380624449604
// MI455X (gfx1250) — matrix operand order repaired and checked by reading, not compiled, not run on hardware
//
#include <hip/hip_runtime.h>
#include <math.h>

// MI455X / gfx1250 fused multi-scale conv kernel.
// bf16 WMMA (v_wmma_f32_16x16x32_bf16) for all GEMMs, f32 elementwise math.

static constexpr int BATCH = 16;
static constexpr int LL = 4096;
static constexpr int CC = 256;
static constexpr int DD = 512;
static constexpr int DP = 128;
static constexpr int DH = 128;
#define EPSV 1e-8f

typedef __attribute__((ext_vector_type(16))) __bf16        v16bf;
typedef __attribute__((ext_vector_type(8)))  float         v8f;
typedef __attribute__((ext_vector_type(8)))  unsigned int  v8u;

__device__ __forceinline__ unsigned short f2bf(float f) {
  unsigned u = __builtin_bit_cast(unsigned, f);
  u += 0x7FFFu + ((u >> 16) & 1u);          // round-to-nearest-even
  return (unsigned short)(u >> 16);
}
__device__ __forceinline__ unsigned packbf(float a, float b) {
  return (unsigned)f2bf(a) | ((unsigned)f2bf(b) << 16);
}
__device__ __forceinline__ float bflo(unsigned u) { return __builtin_bit_cast(float, u << 16); }
__device__ __forceinline__ float bfhi(unsigned u) { return __builtin_bit_cast(float, u & 0xFFFF0000u); }
__device__ __forceinline__ float gelu_f(float v) {
  return 0.5f * v * (1.f + erff(v * 0.70710678118654752440f));
}

struct SM {
  union {
    float    xs[22][CC];        // 22528 B : x halo tile (phase A)
    unsigned fbf[16][DD / 2];   // 16384 B : bf16 feats   (phases B..D)
  };
  union {
    unsigned hbf[4][16][CC / 2]; // 32768 B : bf16 depthwise outputs (A..B)
    float    a1[16][DH];         //  8192 B : attention hidden (C)
  };
  float red[128];
  float attnw[16][4];
};

__global__ void cvt_bf16(const float* __restrict__ src, unsigned short* __restrict__ dst, int n) {
  for (int i = blockIdx.x * blockDim.x + threadIdx.x; i < n; i += gridDim.x * blockDim.x)
    dst[i] = f2bf(src[i]);
}

__global__ __launch_bounds__(256) void fused_msconv(
    const float* __restrict__ x,
    const float* __restrict__ dw1, const float* __restrict__ dw3,
    const float* __restrict__ dw5, const float* __restrict__ dw7,
    const float* __restrict__ cn1, const float* __restrict__ cn3,
    const float* __restrict__ cn5, const float* __restrict__ cn7,
    const unsigned short* __restrict__ pwbf,   // [4][128][256] bf16
    const unsigned short* __restrict__ w1bf,   // [128][512]    bf16
    const float* __restrict__ b1,              // [128]
    const float* __restrict__ w2,              // [4][128]
    const float* __restrict__ b2,              // [4]
    const unsigned short* __restrict__ fwbf,   // [512][512]    bf16
    const unsigned short* __restrict__ rwbf,   // [512][256]    bf16
    const float* __restrict__ nsc,             // [512]
    float* __restrict__ out)                   // [B*L][512]
{
  __shared__ SM sm;
  const int t = threadIdx.x;
  const int lane = t & 31;
  const int wave = t >> 5;
  const int bl0 = blockIdx.x << 4;      // global row base (b*L + l0)
  const int l0 = bl0 & (LL - 1);

  // ---------------- load x halo tile (zero padded at seq edges) --------------
  for (int i = t; i < 22 * CC; i += 256) {
    int r = i >> 8, c = i & 255;
    int l = l0 + r - 3;
    float v = 0.f;
    if (l >= 0 && l < LL) v = x[(size_t)(bl0 - l0 + l) * CC + c];
    sm.xs[r][c] = v;
  }
  __syncthreads();

  // ---------------- depthwise convs, 4 scales, pack bf16 ---------------------
  {
    const int row = t >> 4;
    const int c0 = (t & 15) << 4;
    const float* dws[4] = { dw1, dw3, dw5, dw7 };
#pragma unroll
    for (int s = 0; s < 4; ++s) {
      const int k = 2 * s + 1, hk = s;
      const float* dw = dws[s];
      for (int cc = 0; cc < 16; cc += 2) {
        int c = c0 + cc;
        float v0 = 0.f, v1 = 0.f;
        for (int j = 0; j < k; ++j) {
          v0 += dw[c * k + j]       * sm.xs[row + 3 - hk + j][c];
          v1 += dw[(c + 1) * k + j] * sm.xs[row + 3 - hk + j][c + 1];
        }
        sm.hbf[s][row][c >> 1] = packbf(v0, v1);
      }
    }
  }
  __syncthreads();

  // WMMA fragment helper indices (ISA 16-bit A/B layouts, wave32)
  const int am   = lane & 15;              // A row / D column
  const int mrow = (lane < 16) ? 0 : 8;    // D row offset
  const int aoff = (lane < 16) ? 0 : 4;    // A uint offset within 16-uint k-step
  const int boff = (lane < 16) ? 0 : 8;    // unused; B loads also use aoff

  // ---------------- GEMM1: feats = gelu(rmsnorm(h @ pw^T)) -------------------
  {
    const int s = wave >> 1;               // scale handled by this wave
    v8f acc[4];
#pragma unroll
    for (int ti = 0; ti < 4; ++ti) acc[ti] = {};
    for (int k0 = 0; k0 < CC; k0 += 32) {
      const int u0 = k0 >> 1;
      const unsigned* arow = &sm.hbf[s][am][0];
      v8u au;
#pragma unroll
      for (int j = 0; j < 4; ++j) {
        au[j]     = arow[u0 + aoff + j];
        au[4 + j] = arow[u0 + 8 + aoff + j];
      }
      v16bf af = __builtin_bit_cast(v16bf, au);
#pragma unroll
      for (int ti = 0; ti < 4; ++ti) {
        int n = (((wave & 1) << 2) + ti) * 16 + am;      // 0..127
        const unsigned* wrow = (const unsigned*)pwbf + (size_t)(s * DP + n) * (CC / 2);
        v8u bu;
#pragma unroll
        for (int j = 0; j < 8; ++j) bu[j] = wrow[u0 + aoff + (j & 3) + ((j >> 2) << 3)];
        v16bf bf = __builtin_bit_cast(v16bf, bu);
        acc[ti] = __builtin_amdgcn_wmma_f32_16x16x32_bf16(
            false, af, false, bf, (short)0, acc[ti], false, false);
      }
    }
    // per-(row,scale) sum of squares across 128 channels, cross-lane
    float ssq[8];
#pragma unroll
    for (int r = 0; r < 8; ++r) {
      float v = 0.f;
#pragma unroll
      for (int ti = 0; ti < 4; ++ti) v += acc[ti][r] * acc[ti][r];
      ssq[r] = v;
    }
#pragma unroll
    for (int off = 1; off < 16; off <<= 1) {
#pragma unroll
      for (int r = 0; r < 8; ++r) ssq[r] += __shfl_xor(ssq[r], off, 32);
    }
    if (am == 0) {
#pragma unroll
      for (int r = 0; r < 8; ++r)
        sm.red[(s * 16 + r + mrow) * 2 + (wave & 1)] = ssq[r];
    }
    __syncthreads();

    // normalize, scale by cn, gelu, pack bf16 feats (pairs via lane shuffle)
    const float* cns[4] = { cn1, cn3, cn5, cn7 };
    const float* cn = cns[s];
#pragma unroll
    for (int ti = 0; ti < 4; ++ti) {
      int o = (((wave & 1) << 2) + ti) * 16 + am;
      float cw = cn[o];
#pragma unroll
      for (int r = 0; r < 8; ++r) {
        int row = r + mrow;
        float sum = sm.red[(s * 16 + row) * 2] + sm.red[(s * 16 + row) * 2 + 1];
        float rinv = 1.f / (sqrtf(sum * (1.f / DP)) + EPSV);
        float v = gelu_f(acc[ti][r] * cw * rinv);
        float vn = __shfl_down(v, 1, 32);
        if ((lane & 1) == 0) {
          int col = s * DP + o;                     // even column
          sm.fbf[row][col >> 1] = packbf(v, vn);
        }
      }
    }
  }
  __syncthreads();

  // ---------------- attention hidden: a1 = gelu(feats @ w1^T + b1) -----------
  {
    v8f acc = {};
    int n = wave * 16 + am;
    const unsigned* wrow = (const unsigned*)w1bf + (size_t)n * (DD / 2);
    for (int k0 = 0; k0 < DD; k0 += 32) {
      const int u0 = k0 >> 1;
      const unsigned* arow = &sm.fbf[am][0];
      v8u au, bu;
#pragma unroll
      for (int j = 0; j < 4; ++j) {
        au[j]     = arow[u0 + aoff + j];
        au[4 + j] = arow[u0 + 8 + aoff + j];
      }
#pragma unroll
      for (int j = 0; j < 8; ++j) bu[j] = wrow[u0 + aoff + (j & 3) + ((j >> 2) << 3)];
      acc = __builtin_amdgcn_wmma_f32_16x16x32_bf16(
          false, __builtin_bit_cast(v16bf, au), false, __builtin_bit_cast(v16bf, bu),
          (short)0, acc, false, false);
    }
    float bias = b1[n];
#pragma unroll
    for (int r = 0; r < 8; ++r)
      sm.a1[r + mrow][n] = gelu_f(acc[r] + bias);
  }
  __syncthreads();

  // ---------------- a2 + softmax over 4 scales -------------------------------
  if (t < 64) {
    int row = t >> 2, j = t & 3;
    float a = b2[j];
    const float* wr = w2 + j * DH;
    for (int h = 0; h < DH; ++h) a += wr[h] * sm.a1[row][h];
    sm.attnw[row][j] = a;
  }
  __syncthreads();
  if (t < 16) {
    float s0 = sm.attnw[t][0], s1 = sm.attnw[t][1], s2 = sm.attnw[t][2], s3 = sm.attnw[t][3];
    float m = fmaxf(fmaxf(s0, s1), fmaxf(s2, s3));
    float e0 = expf(s0 - m), e1 = expf(s1 - m), e2 = expf(s2 - m), e3 = expf(s3 - m);
    float inv = 1.f / (e0 + e1 + e2 + e3);
    sm.attnw[t][0] = e0 * inv; sm.attnw[t][1] = e1 * inv;
    sm.attnw[t][2] = e2 * inv; sm.attnw[t][3] = e3 * inv;
  }
  __syncthreads();

  // -------- final: out = rmsnorm((attn*feats)@fw^T + x@rw^T) * nsc -----------
  {
    v8f acc[4];
#pragma unroll
    for (int ti = 0; ti < 4; ++ti) acc[ti] = {};

    // combined GEMM over D=512 (A scaled by softmax weight, group = k/128)
    for (int k0 = 0; k0 < DD; k0 += 32) {
      const int u0 = k0 >> 1;
      float fac = sm.attnw[am][k0 >> 7];
      const unsigned* arow = &sm.fbf[am][0];
      v8u au;
#pragma unroll
      for (int j = 0; j < 4; ++j) {
        unsigned p0 = arow[u0 + aoff + j];
        unsigned p1 = arow[u0 + 8 + aoff + j];
        au[j]     = packbf(bflo(p0) * fac, bfhi(p0) * fac);
        au[4 + j] = packbf(bflo(p1) * fac, bfhi(p1) * fac);
      }
      v16bf af = __builtin_bit_cast(v16bf, au);
#pragma unroll
      for (int ti = 0; ti < 4; ++ti) {
        int n = (wave * 4 + ti) * 16 + am;            // 0..511
        const unsigned* wrow = (const unsigned*)fwbf + (size_t)n * (DD / 2);
        v8u bu;
#pragma unroll
        for (int j = 0; j < 8; ++j) bu[j] = wrow[u0 + aoff + (j & 3) + ((j >> 2) << 3)];
        acc[ti] = __builtin_amdgcn_wmma_f32_16x16x32_bf16(
            false, af, false, __builtin_bit_cast(v16bf, bu), (short)0, acc[ti], false, false);
      }
    }
    // residual GEMM over C=256 (A built from global f32 x)
    const float* xrow = x + (size_t)(bl0 + am) * CC;
    const int kk = (lane < 16) ? 0 : 8;
    for (int k0 = 0; k0 < CC; k0 += 32) {
      v8u au;
#pragma unroll
      for (int j = 0; j < 4; ++j) {
        int ka = k0 + kk + 2 * j;
        int kb = k0 + 16 + kk + 2 * j;
        au[j]     = packbf(xrow[ka], xrow[ka + 1]);
        au[4 + j] = packbf(xrow[kb], xrow[kb + 1]);
      }
      v16bf af = __builtin_bit_cast(v16bf, au);
      const int u0 = k0 >> 1;
#pragma unroll
      for (int ti = 0; ti < 4; ++ti) {
        int n = (wave * 4 + ti) * 16 + am;
        const unsigned* wrow = (const unsigned*)rwbf + (size_t)n * (CC / 2);
        v8u bu;
#pragma unroll
        for (int j = 0; j < 8; ++j) bu[j] = wrow[u0 + aoff + (j & 3) + ((j >> 2) << 3)];
        acc[ti] = __builtin_amdgcn_wmma_f32_16x16x32_bf16(
            false, af, false, __builtin_bit_cast(v16bf, bu), (short)0, acc[ti], false, false);
      }
    }

    // final RMSNorm over 512: cross-lane + cross-wave reduction
    float ssq[8];
#pragma unroll
    for (int r = 0; r < 8; ++r) {
      float v = 0.f;
#pragma unroll
      for (int ti = 0; ti < 4; ++ti) v += acc[ti][r] * acc[ti][r];
      ssq[r] = v;
    }
#pragma unroll
    for (int off = 1; off < 16; off <<= 1) {
#pragma unroll
      for (int r = 0; r < 8; ++r) ssq[r] += __shfl_xor(ssq[r], off, 32);
    }
    if (am == 0) {
#pragma unroll
      for (int r = 0; r < 8; ++r) sm.red[(r + mrow) * 8 + wave] = ssq[r];
    }
    __syncthreads();

#pragma unroll
    for (int r = 0; r < 8; ++r) {
      int row = r + mrow;
      float sum = 0.f;
#pragma unroll
      for (int w = 0; w < 8; ++w) sum += sm.red[row * 8 + w];
      float rinv = 1.f / (sqrtf(sum * (1.f / DD)) + EPSV);
      float* orow = out + (size_t)(bl0 + row) * DD;
#pragma unroll
      for (int ti = 0; ti < 4; ++ti) {
        int col = (wave * 4 + ti) * 16 + am;
        orow[col] = acc[ti][r] * nsc[col] * rinv;
      }
    }
  }
}

extern "C" void kernel_launch(void* const* d_in, const int* in_sizes, int n_in,
                              void* d_out, int out_size, void* d_ws, size_t ws_size,
                              hipStream_t stream) {
  (void)in_sizes; (void)n_in; (void)out_size; (void)ws_size;
  const float* x   = (const float*)d_in[0];
  const float* dw1 = (const float*)d_in[1];
  const float* pw1 = (const float*)d_in[2];
  const float* cn1 = (const float*)d_in[3];
  const float* dw3 = (const float*)d_in[4];
  const float* pw3 = (const float*)d_in[5];
  const float* cn3 = (const float*)d_in[6];
  const float* dw5 = (const float*)d_in[7];
  const float* pw5 = (const float*)d_in[8];
  const float* cn5 = (const float*)d_in[9];
  const float* dw7 = (const float*)d_in[10];
  const float* pw7 = (const float*)d_in[11];
  const float* cn7 = (const float*)d_in[12];
  const float* w1  = (const float*)d_in[13];
  const float* b1  = (const float*)d_in[14];
  const float* w2  = (const float*)d_in[15];
  const float* b2  = (const float*)d_in[16];
  const float* fw  = (const float*)d_in[17];
  const float* nsc = (const float*)d_in[18];
  const float* rw  = (const float*)d_in[19];

  unsigned short* pwbf = (unsigned short*)d_ws;     // 4*128*256
  unsigned short* w1bf = pwbf + 4 * DP * CC;        // 128*512
  unsigned short* fwbf = w1bf + DH * DD;            // 512*512
  unsigned short* rwbf = fwbf + DD * DD;            // 512*256

  cvt_bf16<<<128, 256, 0, stream>>>(pw1, pwbf + 0 * DP * CC, DP * CC);
  cvt_bf16<<<128, 256, 0, stream>>>(pw3, pwbf + 1 * DP * CC, DP * CC);
  cvt_bf16<<<128, 256, 0, stream>>>(pw5, pwbf + 2 * DP * CC, DP * CC);
  cvt_bf16<<<128, 256, 0, stream>>>(pw7, pwbf + 3 * DP * CC, DP * CC);
  cvt_bf16<<<128, 256, 0, stream>>>(w1, w1bf, DH * DD);
  cvt_bf16<<<512, 256, 0, stream>>>(fw, fwbf, DD * DD);
  cvt_bf16<<<256, 256, 0, stream>>>(rw, rwbf, DD * CC);

  fused_msconv<<<(BATCH * LL) / 16, 256, 0, stream>>>(
      x, dw1, dw3, dw5, dw7, cn1, cn3, cn5, cn7,
      pwbf, w1bf, b1, w2, b2, fwbf, rwbf, nsc, (float*)d_out);
}
